// CausalSelectiveSelfAttentionWithMemoryPenalty_19859928776997
// MI455X (gfx1250) — matrix operand order repaired and checked by reading, not compiled, not run on hardware
//
#include <hip/hip_runtime.h>

// ---- problem constants ----
#define Bz   2
#define Tz   2048
#define Cz   768
#define Hz   12
#define HDz  64
#define NQKV 2304
#define TAUz 1.0f

typedef __attribute__((ext_vector_type(16))) __bf16       v16bf;
typedef __attribute__((ext_vector_type(8)))  float        v8f;
typedef __attribute__((ext_vector_type(8)))  unsigned int v8u;

static __device__ __forceinline__ unsigned short f2bf(float f) {
  union { float f; unsigned u; } c; c.f = f;
  unsigned u = c.u;
  u += 0x7FFFu + ((u >> 16) & 1u);          // round-to-nearest-even
  return (unsigned short)(u >> 16);
}

// A fragment: 16x32 bf16 from row-major ushort source, row stride ld (elements).
// Layout (CDNA5 ISA 7.12.2): lane m=lane&15; vgpr j packs K = (j&3)*2 + half*8 + (j>=4?16:0), K+1
static __device__ __forceinline__ v16bf load_a_bf16(const unsigned short* base, int ld) {
  const int lane = threadIdx.x & 31;
  const int m = lane & 15, hh = lane >> 4;
  v8u a;
#pragma unroll
  for (int j = 0; j < 8; ++j) {
    int k = ((j & 3) * 2) + hh * 8 + ((j & 4) ? 16 : 0);
    a[j] = *(const unsigned int*)(base + m * ld + k);
  }
  return __builtin_bit_cast(v16bf, a);
}

// B fragment 32x16 where element (k,n) = src[n*ld + k]  (pairs contiguous in k).
// Layout: lane n=lane&15; vgpr j packs K = (j&3)*2 + half*8 + (j>=4?16:0), K+1
static __device__ __forceinline__ v16bf load_b_bf16_T(const unsigned short* base, int ld) {
  const int lane = threadIdx.x & 31;
  const int n = lane & 15, hh = lane >> 4;
  v8u b;
#pragma unroll
  for (int j = 0; j < 8; ++j) {
    int k = ((j & 3) * 2) + hh * 8 + ((j & 4) ? 16 : 0);
    b[j] = *(const unsigned int*)(base + n * ld + k);
  }
  return __builtin_bit_cast(v16bf, b);
}

// ---------------- Kernel 0a: fp32 -> bf16 elementwise convert (x activations) ----------------
__global__ __launch_bounds__(256) void convert_bf16_kernel(
    const float* __restrict__ src, unsigned short* __restrict__ dst, int n4) {
  int i = (blockIdx.x * 256 + threadIdx.x) * 4;    // 4 elements/thread
  if (i < n4) {
#pragma unroll
    for (int j = 0; j < 4; ++j) dst[i + j] = f2bf(src[i + j]);
  }
}

// ---------------- Kernel 0b: fp32 [K x N] -> bf16 [N x K] weight transpose ----------------
// 32x32 tile through LDS (33 pitch: bank-conflict-free), 256 threads = 32x8.
__global__ __launch_bounds__(256) void transpose_bf16_kernel(
    const float* __restrict__ W, unsigned short* __restrict__ Wt, int K, int N) {
  __shared__ unsigned short tile[32][33];
  const int n0 = blockIdx.x * 32, k0 = blockIdx.y * 32;
  const int tx = threadIdx.x & 31, ty = threadIdx.x >> 5;   // 32 cols x 8 rows
#pragma unroll
  for (int i = 0; i < 32; i += 8)
    tile[ty + i][tx] = f2bf(W[(size_t)(k0 + ty + i) * N + n0 + tx]);
  __syncthreads();
#pragma unroll
  for (int i = 0; i < 32; i += 8)
    Wt[(size_t)(n0 + ty + i) * K + k0 + tx] = tile[tx][ty + i];
}

// ---------------- Kernel 1: QKV GEMM  [4096 x 768] x [768 x 2304] ----------------
// 8 waves/block, wave-per-16x16 tile; A from pre-converted bf16 x, B from pre-transposed
// bf16 weights -- inner loop is pure b128 loads + wmma.
// Scatters q -> [B,H,T,HD], k -> [B,H,T,HD], v -> [B,H,HD,T] (transposed), all bf16.
__global__ __launch_bounds__(256) void qkv_gemm_kernel(
    const unsigned short* __restrict__ xb, const unsigned short* __restrict__ WqkvT,
    const float* __restrict__ bqkv,
    unsigned short* __restrict__ qb, unsigned short* __restrict__ kb,
    unsigned short* __restrict__ vtb) {
  const int wave  = threadIdx.x >> 5;
  const int lane  = threadIdx.x & 31;
  const int tileM = blockIdx.y;              // 0..255
  const int tileN = blockIdx.x * 8 + wave;   // 0..143
  const int row0  = tileM * 16;
  const int col0  = tileN * 16;

  v8f acc = {};
#pragma unroll 2
  for (int kc = 0; kc < Cz; kc += 32) {
    v16bf a = load_a_bf16  (xb    + (size_t)row0 * Cz + kc, Cz);
    v16bf b = load_b_bf16_T(WqkvT + (size_t)col0 * Cz + kc, Cz);
    acc = __builtin_amdgcn_wmma_f32_16x16x32_bf16(false, a, false, b, (short)0, acc, false, false);
    if (kc + 32 < Cz)
      __builtin_prefetch(WqkvT + (size_t)col0 * Cz + kc + 32, 0, 1);
  }

  const int m0 = (lane >> 4) * 8;
  const int cn = lane & 15;
#pragma unroll
  for (int r = 0; r < 8; ++r) {
    int row = row0 + m0 + r;                 // = b*T + t
    int col = col0 + cn;                     // 0..2303
    float v = acc[r] + bqkv[col];
    int sec    = col / Cz;                   // 0:q 1:k 2:v
    int within = col - sec * Cz;
    int hh = within >> 6, dd = within & 63;
    int bb = row >> 11,   tt = row & (Tz - 1);
    unsigned short bv = f2bf(v);
    if (sec == 0)      qb [(((size_t)(bb * Hz + hh) * Tz + tt) << 6) + dd] = bv;
    else if (sec == 1) kb [(((size_t)(bb * Hz + hh) * Tz + tt) << 6) + dd] = bv;
    else               vtb[(((size_t)(bb * Hz + hh) << 6) + dd) * Tz + tt] = bv;
  }
}

// ---------------- Kernel 2a: head-0 scores S = mask(relu(QK^T/8)) -> FF buffer ----------------
__global__ __launch_bounds__(256) void head0_scores_kernel(
    const unsigned short* __restrict__ qb, const unsigned short* __restrict__ kb,
    float* __restrict__ S) {
  const int wave  = threadIdx.x >> 5;
  const int lane  = threadIdx.x & 31;
  const int ktile = blockIdx.x * 8 + wave;   // 0..127
  const int qtile = blockIdx.y;              // 0..127
  const int bb    = blockIdx.z;
  const int q0 = qtile * 16, k0 = ktile * 16;
  float* Sp = S + (size_t)bb * Tz * Tz;
  const int m0 = (lane >> 4) * 8;
  const int cn = lane & 15;

  if (qtile < ktile) {                       // strictly above diagonal: all zero
#pragma unroll
    for (int r = 0; r < 8; ++r) Sp[(size_t)(q0 + m0 + r) * Tz + k0 + cn] = 0.f;
    return;
  }
  const unsigned short* qh = qb + (size_t)(bb * Hz) * Tz * HDz;   // head 0
  const unsigned short* kh = kb + (size_t)(bb * Hz) * Tz * HDz;
  v8f acc = {};
#pragma unroll
  for (int kc = 0; kc < HDz; kc += 32) {
    v16bf a = load_a_bf16  (qh + (size_t)q0 * HDz + kc, HDz);
    v16bf b = load_b_bf16_T(kh + (size_t)k0 * HDz + kc, HDz);
    acc = __builtin_amdgcn_wmma_f32_16x16x32_bf16(false, a, false, b, (short)0, acc, false, false);
  }
#pragma unroll
  for (int r = 0; r < 8; ++r) {
    int qg = q0 + m0 + r, kg = k0 + cn;
    float s = acc[r] * 0.125f;
    s = (qg > kg && kg > 0) ? fmaxf(s, 0.f) : 0.f;   // causal-strict + protect col 0 + no diag
    Sp[(size_t)qg * Tz + kg] = s;
  }
}

// ---------------- Kernel 2b: in-place exclusive cumsum over query dim (per key column) ----------------
__global__ __launch_bounds__(256) void ff_cumsum_kernel(float* __restrict__ FF) {
  int gid  = blockIdx.x * 256 + threadIdx.x;   // 0..B*T-1
  int bb   = gid >> 11;
  int kcol = gid & (Tz - 1);
  float* Fp = FF + (size_t)bb * Tz * Tz + kcol;
  float run = 0.f;
  for (int q = 0; q < Tz; ++q) {
    float s = Fp[(size_t)q * Tz];
    Fp[(size_t)q * Tz] = run;                  // FF[q] = sum_{q'<q} S[q']
    run += s;
  }
}

// ---------------- Kernel 2c: FF_sum[b,q] = sum_k min(FF[b,q,k], TAU) ----------------
__global__ __launch_bounds__(256) void ff_rowsum_kernel(const float* __restrict__ FF,
                                                        float* __restrict__ FFsum) {
  __shared__ float red[256];
  const int row = blockIdx.x;                  // 0..B*T-1
  const int bb  = row >> 11;
  const float* Fp = FF + (size_t)bb * Tz * Tz + (size_t)(row & (Tz - 1)) * Tz;
  float s = 0.f;
  for (int k = threadIdx.x; k < Tz; k += 256) s += fminf(Fp[k], TAUz);
  red[threadIdx.x] = s;
  __syncthreads();
  for (int st = 128; st > 0; st >>= 1) {
    if (threadIdx.x < st) red[threadIdx.x] += red[threadIdx.x + st];
    __syncthreads();
  }
  if (threadIdx.x == 0) FFsum[row] = red[0];
}

// ---------------- Kernel 2d: M[0,b,i,j] = i - FF_sum[b,j]/TAU ----------------
__global__ __launch_bounds__(256) void m_out_kernel(const float* __restrict__ FFsum,
                                                    float* __restrict__ Mout) {
  size_t idx = (size_t)blockIdx.x * 256 + threadIdx.x;   // < B*T*T
  int    j   = (int)(idx & (Tz - 1));
  size_t t   = idx >> 11;
  int    i   = (int)(t & (Tz - 1));
  int    bb  = (int)(t >> 11);
  Mout[idx] = (float)i - FFsum[bb * Tz + j] * (1.0f / TAUz);
}

// ---------------- Kernel 3: flash attention with FF bias, WMMA QK^T and P*V ----------------
__global__ __launch_bounds__(256) void attn_kernel(
    const unsigned short* __restrict__ qb, const unsigned short* __restrict__ kb,
    const unsigned short* __restrict__ vtb, const float* __restrict__ FF,
    unsigned short* __restrict__ ah) {
  __shared__ unsigned short ldsP[8 * 16 * 32];           // per-wave 16x32 bf16 P tile
  const int wave = threadIdx.x >> 5;
  const int lane = threadIdx.x & 31;
  unsigned short* P = ldsP + wave * (16 * 32);
  const int qtile = blockIdx.x * 8 + wave;               // 0..127
  const int hh = blockIdx.y, bb = blockIdx.z;
  const int q0 = qtile * 16;
  const unsigned short* qh = qb  + (size_t)(bb * Hz + hh) * Tz * HDz;
  const unsigned short* kh = kb  + (size_t)(bb * Hz + hh) * Tz * HDz;
  const unsigned short* vh = vtb + (size_t)(bb * Hz + hh) * HDz * Tz;  // [HD, T]
  const float* Fp = FF + (size_t)bb * Tz * Tz;

  const int m0 = (lane >> 4) * 8;
  const int cn = lane & 15;

  v8f o0 = {}, o1 = {}, o2 = {}, o3 = {};
  float mrow[8], lrow[8];
#pragma unroll
  for (int r = 0; r < 8; ++r) { mrow[r] = -3.0e38f; lrow[r] = 0.f; }

  const v16bf aq0 = load_a_bf16(qh + (size_t)q0 * HDz + 0,  HDz);
  const v16bf aq1 = load_a_bf16(qh + (size_t)q0 * HDz + 32, HDz);

  const int nchunks = (q0 >> 5) + 1;                     // 32-key chunks covering causal extent
  for (int ch = 0; ch < nchunks; ++ch) {
    const int kbase = ch * 32;
    v8f s0 = {}, s1 = {};
    {
      v16bf b00 = load_b_bf16_T(kh + (size_t)(kbase +  0) * HDz + 0,  HDz);
      v16bf b01 = load_b_bf16_T(kh + (size_t)(kbase +  0) * HDz + 32, HDz);
      s0 = __builtin_amdgcn_wmma_f32_16x16x32_bf16(false, aq0, false, b00, (short)0, s0, false, false);
      s0 = __builtin_amdgcn_wmma_f32_16x16x32_bf16(false, aq1, false, b01, (short)0, s0, false, false);
      v16bf b10 = load_b_bf16_T(kh + (size_t)(kbase + 16) * HDz + 0,  HDz);
      v16bf b11 = load_b_bf16_T(kh + (size_t)(kbase + 16) * HDz + 32, HDz);
      s1 = __builtin_amdgcn_wmma_f32_16x16x32_bf16(false, aq0, false, b10, (short)0, s1, false, false);
      s1 = __builtin_amdgcn_wmma_f32_16x16x32_bf16(false, aq1, false, b11, (short)0, s1, false, false);
    }
    // scale + FF bias + causal mask (element (q0+m0+r, kbase[+16]+cn))
    float sv0[8], sv1[8];
#pragma unroll
    for (int r = 0; r < 8; ++r) {
      int qg  = q0 + m0 + r;
      int k0g = kbase + cn, k1g = kbase + 16 + cn;
      float a = s0[r] * 0.125f - Fp[(size_t)qg * Tz + k0g];
      float c = s1[r] * 0.125f - Fp[(size_t)qg * Tz + k1g];
      if (k0g > qg) a = -3.0e38f;
      if (k1g > qg) c = -3.0e38f;
      sv0[r] = a; sv1[r] = c;
    }
    // online softmax; row stats replicated across each 16-lane half
#pragma unroll
    for (int r = 0; r < 8; ++r) {
      float cmax = fmaxf(sv0[r], sv1[r]);
      cmax = fmaxf(cmax, __shfl_xor(cmax, 1, 32));
      cmax = fmaxf(cmax, __shfl_xor(cmax, 2, 32));
      cmax = fmaxf(cmax, __shfl_xor(cmax, 4, 32));
      cmax = fmaxf(cmax, __shfl_xor(cmax, 8, 32));
      float mnew  = fmaxf(mrow[r], cmax);
      float scale = __expf(mrow[r] - mnew);
      float p0 = __expf(sv0[r] - mnew);
      float p1 = __expf(sv1[r] - mnew);
      float ps = p0 + p1;
      ps += __shfl_xor(ps, 1, 32);
      ps += __shfl_xor(ps, 2, 32);
      ps += __shfl_xor(ps, 4, 32);
      ps += __shfl_xor(ps, 8, 32);
      lrow[r] = lrow[r] * scale + ps;
      mrow[r] = mnew;
      o0[r] *= scale; o1[r] *= scale; o2[r] *= scale; o3[r] *= scale;
      int m = m0 + r;
      P[m * 32 + cn]      = f2bf(p0);        // C-layout -> row-major bf16 in LDS
      P[m * 32 + 16 + cn] = f2bf(p1);
    }
    asm volatile("s_wait_dscnt 0" ::: "memory");         // P stores visible before reload
    v16bf ap  = load_a_bf16(P, 32);                      // 16x32 A fragment (K = 32 keys)
    v16bf bv0 = load_b_bf16_T(vh + (size_t)( 0) * Tz + kbase, Tz);
    v16bf bv1 = load_b_bf16_T(vh + (size_t)(16) * Tz + kbase, Tz);
    v16bf bv2 = load_b_bf16_T(vh + (size_t)(32) * Tz + kbase, Tz);
    v16bf bv3 = load_b_bf16_T(vh + (size_t)(48) * Tz + kbase, Tz);
    o0 = __builtin_amdgcn_wmma_f32_16x16x32_bf16(false, ap, false, bv0, (short)0, o0, false, false);
    o1 = __builtin_amdgcn_wmma_f32_16x16x32_bf16(false, ap, false, bv1, (short)0, o1, false, false);
    o2 = __builtin_amdgcn_wmma_f32_16x16x32_bf16(false, ap, false, bv2, (short)0, o2, false, false);
    o3 = __builtin_amdgcn_wmma_f32_16x16x32_bf16(false, ap, false, bv3, (short)0, o3, false, false);
  }

#pragma unroll
  for (int r = 0; r < 8; ++r) {
    float inv = 1.0f / lrow[r];
    int t = q0 + m0 + r;
    size_t rowoff = ((size_t)bb * Tz + t) * (Hz * HDz) + hh * HDz;
    ah[rowoff +  0 + cn] = f2bf(o0[r] * inv);
    ah[rowoff + 16 + cn] = f2bf(o1[r] * inv);
    ah[rowoff + 32 + cn] = f2bf(o2[r] * inv);
    ah[rowoff + 48 + cn] = f2bf(o3[r] * inv);
  }
}

// ---------------- Kernel 4: output projection [4096 x 768] x [768 x 768] + bias -> f32 ----------------
__global__ __launch_bounds__(256) void proj_gemm_kernel(
    const unsigned short* __restrict__ ah, const unsigned short* __restrict__ WprojT,
    const float* __restrict__ bproj, float* __restrict__ yout) {
  const int wave  = threadIdx.x >> 5;
  const int lane  = threadIdx.x & 31;
  const int tileM = blockIdx.y;              // 0..255
  const int tileN = blockIdx.x * 8 + wave;   // 0..47
  const int row0  = tileM * 16, col0 = tileN * 16;
  v8f acc = {};
#pragma unroll 4
  for (int kc = 0; kc < Cz; kc += 32) {
    v16bf a = load_a_bf16  (ah     + (size_t)row0 * Cz + kc, Cz);
    v16bf b = load_b_bf16_T(WprojT + (size_t)col0 * Cz + kc, Cz);
    acc = __builtin_amdgcn_wmma_f32_16x16x32_bf16(false, a, false, b, (short)0, acc, false, false);
    if (kc + 32 < Cz)
      __builtin_prefetch(WprojT + (size_t)col0 * Cz + kc + 32, 0, 1);
  }
  const int m0 = (lane >> 4) * 8, cn = lane & 15;
#pragma unroll
  for (int r = 0; r < 8; ++r) {
    int row = row0 + m0 + r, col = col0 + cn;
    yout[(size_t)row * Cz + col] = acc[r] + bproj[col];
  }
}

// ---------------- launch ----------------
extern "C" void kernel_launch(void* const* d_in, const int* in_sizes, int n_in,
                              void* d_out, int out_size, void* d_ws, size_t ws_size,
                              hipStream_t stream) {
  (void)in_sizes; (void)n_in; (void)out_size; (void)ws_size;
  const float* x     = (const float*)d_in[0];
  const float* Wqkv  = (const float*)d_in[1];
  const float* bqkv  = (const float*)d_in[2];
  const float* Wproj = (const float*)d_in[3];
  const float* bproj = (const float*)d_in[4];

  float* y    = (float*)d_out;                               // [B,T,C]
  float* Mout = y + (size_t)Bz * Tz * Cz;                    // [1,B,T,T]

  char* ws = (char*)d_ws;                                    // byte offsets, 256-aligned
  unsigned short* qb     = (unsigned short*)(ws);            //  6.29 MB  q  [B,H,T,HD] bf16
  unsigned short* kb     = (unsigned short*)(ws +  6291456); //  6.29 MB  k  [B,H,T,HD] bf16
  unsigned short* vtb    = (unsigned short*)(ws + 12582912); //  6.29 MB  v^T[B,H,HD,T] bf16
  unsigned short* ahbuf  = (unsigned short*)(ws + 18874368); //  6.29 MB  attn-out [B,T,768] bf16
  float*          FF     = (float*)        (ws + 25165824);  // 33.55 MB  S/FF [B,T,T] f32
  float*          FFsum  = (float*)        (ws + 58720256);  // 16 KB     [B,T] f32
  unsigned short* WqkvT  = (unsigned short*)(ws + 58736640); //  3.54 MB  Wqkv^T [2304 x 768] bf16
  unsigned short* WprojT = (unsigned short*)(ws + 62275584); //  1.18 MB  Wproj^T [768 x 768] bf16
  unsigned short* xb     = (unsigned short*)(ws + 63455232); //  6.29 MB  x bf16 [B,T,C]

  dim3 blk(256);
  const int xn = Bz * Tz * Cz;                               // 3,145,728 (mult of 1024)
  convert_bf16_kernel  <<<dim3(xn / 1024), blk, 0, stream>>>(x, xb, xn);
  transpose_bf16_kernel<<<dim3(NQKV / 32, Cz / 32, 1), blk, 0, stream>>>(Wqkv,  WqkvT,  Cz, NQKV);
  transpose_bf16_kernel<<<dim3(Cz / 32,   Cz / 32, 1), blk, 0, stream>>>(Wproj, WprojT, Cz, Cz);
  qkv_gemm_kernel    <<<dim3(18, 256, 1), blk, 0, stream>>>(xb, WqkvT, bqkv, qb, kb, vtb);
  head0_scores_kernel<<<dim3(16, 128, Bz), blk, 0, stream>>>(qb, kb, FF);
  ff_cumsum_kernel   <<<dim3((Bz * Tz) / 256), blk, 0, stream>>>(FF);
  ff_rowsum_kernel   <<<dim3(Bz * Tz), blk, 0, stream>>>(FF, FFsum);
  m_out_kernel       <<<dim3((unsigned)(((size_t)Bz * Tz * Tz) / 256)), blk, 0, stream>>>(FFsum, Mout);
  attn_kernel        <<<dim3(16, Hz, Bz), blk, 0, stream>>>(qb, kb, vtb, FF, ahbuf);
  proj_gemm_kernel   <<<dim3(6, 256, 1), blk, 0, stream>>>(ahbuf, WprojT, bproj, y);
}
